// SingleHeadHyperbolicAttention_77893526880293
// MI455X (gfx1250) — matrix operand order repaired and checked by reading, not compiled, not run on hardware
//
#include <hip/hip_runtime.h>
#include <math.h>

typedef _Float16 v16h __attribute__((ext_vector_type(16)));
typedef _Float16 h8   __attribute__((ext_vector_type(8)));
typedef _Float16 h4   __attribute__((ext_vector_type(4)));
typedef float    v8f  __attribute__((ext_vector_type(8)));
typedef float    f4   __attribute__((ext_vector_type(4)));

static constexpr int BB = 4;
static constexpr int SS = 2048;
static constexpr int DD = 512;
#define EPSF 1e-8f

__device__ inline v16h pack16(h8 lo, h8 hi2) {
  v16h r;
#pragma unroll
  for (int j = 0; j < 8; ++j) { r[j] = lo[j]; r[j + 8] = hi2[j]; }
  return r;
}

// ---------------------------------------------------------------------------
// Kernel 1: per-row norms of q,k plus log-map scale for v; emit f16 q,k.
// One 256-thread block per (b,row).
// ---------------------------------------------------------------------------
__global__ void prep_kernel(const float* __restrict__ q, const float* __restrict__ k,
                            const float* __restrict__ v,
                            _Float16* __restrict__ qh, _Float16* __restrict__ kh,
                            float* __restrict__ qsq, float* __restrict__ ksq,
                            float* __restrict__ vscale) {
  const int row = blockIdx.x;            // 0 .. B*S-1
  const int tid = threadIdx.x;           // 256
  const size_t off = (size_t)row * DD;

  __shared__ float rq[256], rk[256], rv[256];
  float sq = 0.f, sk = 0.f, sv = 0.f;
  for (int j = tid; j < DD; j += 256) {
    float a = q[off + j], b = k[off + j], c = v[off + j];
    sq += a * a; sk += b * b; sv += c * c;
  }
  rq[tid] = sq; rk[tid] = sk; rv[tid] = sv;
  __syncthreads();
  for (int s2 = 128; s2 > 0; s2 >>= 1) {
    if (tid < s2) {
      rq[tid] += rq[tid + s2];
      rk[tid] += rk[tid + s2];
      rv[tid] += rv[tid + s2];
    }
    __syncthreads();
  }
  if (tid == 0) {
    qsq[row] = rq[0];
    ksq[row] = rk[0];
    float n    = sqrtf(rv[0]);
    float nmax = fmaxf(n, EPSF);
    float ncl  = fminf(nmax, 1.0f - EPSF);
    vscale[row] = atanhf(ncl) / nmax;    // log-map-at-origin scale
  }
  for (int j = tid; j < DD; j += 256) {
    qh[off + j] = (_Float16)q[off + j];
    kh[off + j] = (_Float16)k[off + j];
  }
}

// ---------------------------------------------------------------------------
// Kernel 2: build v_tan TRANSPOSED: vtT[b][d][t] = (f16)(v[b][t][d]*vscale[b][t])
// LDS-tiled 32x32 transpose; coalesced read and write.
// ---------------------------------------------------------------------------
__global__ void vtan_transpose_kernel(const float* __restrict__ v,
                                      const float* __restrict__ vscale,
                                      _Float16* __restrict__ vtT) {
  const int nTt = SS / 32;               // 64
  const int nTd = DD / 32;               // 16
  const int tile = blockIdx.x;
  const int b   = tile / (nTt * nTd);
  const int rem = tile % (nTt * nTd);
  const int ti  = rem / nTd, di = rem % nTd;
  const int t0 = ti * 32, d0 = di * 32;
  const int tid = threadIdx.x;

  __shared__ _Float16 lds[32][33];       // pad to dodge bank conflicts

  // read: thread (ty=tid/8 row t, tx=tid%8 group of 4 d)
  {
    const int ty = tid >> 3, tx = tid & 7;
    const int t  = t0 + ty;
    const float vs = vscale[(size_t)b * SS + t];
    f4 x = *(const f4*)(v + ((size_t)b * SS + t) * DD + d0 + tx * 4);
#pragma unroll
    for (int j = 0; j < 4; ++j) lds[ty][tx * 4 + j] = (_Float16)(x[j] * vs);
  }
  __syncthreads();
  // write transposed: thread (dy=tid/8 row d, txw=tid%8 group of 4 t)
  {
    const int dy = tid >> 3, txw = tid & 7;
    h4 y;
#pragma unroll
    for (int j = 0; j < 4; ++j) y[j] = lds[txw * 4 + j][dy];
    *(h4*)(vtT + ((size_t)b * DD + d0 + dy) * SS + t0 + txw * 4) = y;
  }
}

// ---------------------------------------------------------------------------
// Kernel 3: scores. Each wave: one 16-row s-tile x FOUR 16-col t-tiles
// (A fragment reused across 4 WMMAs; 8 waves/block share the same B t-tiles
// so k operands hit WGP$). Then Poincare distance -> score, written to attn.
// ---------------------------------------------------------------------------
__global__ void scores_kernel(const _Float16* __restrict__ qh,
                              const _Float16* __restrict__ kh,
                              const float* __restrict__ qsq,
                              const float* __restrict__ ksq,
                              const float* __restrict__ beta,
                              const float* __restrict__ cofs,
                              float* __restrict__ attn) {
  const int lane = threadIdx.x & 31;
  const int wave = threadIdx.x >> 5;
  const int nSg = (SS / 16) / 8;         // 16 s-groups (8 s-tiles per block)
  const int nTg = (SS / 16) / 4;         // 32 t-groups (4 t-tiles per wave)
  const int bid = blockIdx.x;
  const int b   = bid / (nSg * nTg);
  const int rem = bid % (nSg * nTg);
  const int sg  = rem / nTg, tg = rem % nTg;
  const int srow = (sg * 8 + wave) * 16;
  const int t0   = tg * 64;
  const int laneM = lane & 15;
  const int hi    = lane >> 4;

  const float lam = log1pf(expf(beta[0]));   // softplus(beta)
  const float cc  = cofs[0];

  const _Float16* qbase = qh + ((size_t)b * SS + srow + laneM) * DD;
  const _Float16* kb[4];
#pragma unroll
  for (int tt = 0; tt < 4; ++tt)
    kb[tt] = kh + ((size_t)b * SS + t0 + tt * 16 + laneM) * DD;

  v8f acc[4] = {};
  for (int k0 = 0; k0 < DD; k0 += 32) {
    h8 a0 = *(const h8*)(qbase + k0 + hi * 8);
    h8 a1 = *(const h8*)(qbase + k0 + hi * 8 + 16);
    const v16h A = pack16(a0, a1);
#pragma unroll
    for (int tt = 0; tt < 4; ++tt) {
      h8 b0 = *(const h8*)(kb[tt] + k0 + hi * 8);
      h8 b1 = *(const h8*)(kb[tt] + k0 + hi * 8 + 16);
      const v16h Bm = pack16(b0, b1);
      acc[tt] = __builtin_amdgcn_wmma_f32_16x16x32_f16(
          false, A, false, Bm, (short)0, acc[tt], false, false);
    }
  }

  // C layout: vgpr r, lane L -> row M = r + (L>=16 ? 8 : 0), col N = L%16
  float xsqv[8];
#pragma unroll
  for (int r = 0; r < 8; ++r)
    xsqv[r] = qsq[(size_t)b * SS + srow + r + hi * 8];

#pragma unroll
  for (int tt = 0; tt < 4; ++tt) {
    const int trow = t0 + tt * 16;
    const float ysq = ksq[(size_t)b * SS + trow + laneM];
    float* ocol = attn + (size_t)b * SS * SS + trow + laneM;
#pragma unroll
    for (int r = 0; r < 8; ++r) {
      const int M = r + hi * 8;
      const float xsq  = xsqv[r];
      const float xy   = acc[tt][r];
      const float diff = xsq - 2.f * xy + ysq;
      const float den  = fmaxf((1.f - xsq) * (1.f - ysq), EPSF);
      const float arg  = fmaxf(1.f + 2.f * diff / den, 1.f + EPSF);
      const float dist = acoshf(arg);
      ocol[(size_t)(srow + M) * SS] = -lam * dist - cc;
    }
  }
}

// ---------------------------------------------------------------------------
// Kernel 4: row softmax over attn (in place). One block per (b,s) row.
// ---------------------------------------------------------------------------
__global__ void softmax_kernel(float* __restrict__ attn) {
  const int row = blockIdx.x;
  const int tid = threadIdx.x;
  float* p = attn + (size_t)row * SS;

  float vals[8];
  float m = -INFINITY;
#pragma unroll
  for (int j = 0; j < 8; ++j) { vals[j] = p[j * 256 + tid]; m = fmaxf(m, vals[j]); }

  __shared__ float red[256];
  red[tid] = m; __syncthreads();
  for (int s2 = 128; s2 > 0; s2 >>= 1) {
    if (tid < s2) red[tid] = fmaxf(red[tid], red[tid + s2]);
    __syncthreads();
  }
  m = red[0];
  __syncthreads();

  float sum = 0.f;
#pragma unroll
  for (int j = 0; j < 8; ++j) { vals[j] = __expf(vals[j] - m); sum += vals[j]; }
  red[tid] = sum; __syncthreads();
  for (int s2 = 128; s2 > 0; s2 >>= 1) {
    if (tid < s2) red[tid] += red[tid + s2];
    __syncthreads();
  }
  const float inv = 1.f / red[0];
#pragma unroll
  for (int j = 0; j < 8; ++j) p[j * 256 + tid] = vals[j] * inv;
}

// ---------------------------------------------------------------------------
// Kernel 5: out_tan = attn @ v_tan. Each wave: one s-tile x FOUR d-tiles.
// A = attn (f32 -> f16 on the fly, amortized over 4 WMMAs); B = vtT with
// contiguous 16-byte loads thanks to the transposed layout.
// ---------------------------------------------------------------------------
__global__ void av_kernel(const float* __restrict__ attn,
                          const _Float16* __restrict__ vtT,
                          float* __restrict__ out) {
  const int lane = threadIdx.x & 31;
  const int wave = threadIdx.x >> 5;
  const int nSg = (SS / 16) / 8;         // 16
  const int nDg = (DD / 16) / 4;         // 8
  const int bid = blockIdx.x;
  const int b   = bid / (nSg * nDg);
  const int rem = bid % (nSg * nDg);
  const int sg  = rem / nDg, dg = rem % nDg;
  const int srow = (sg * 8 + wave) * 16;
  const int d0   = dg * 64;
  const int laneM = lane & 15;
  const int hi    = lane >> 4;

  const float* arow = attn + ((size_t)b * SS + srow + laneM) * SS;
  const _Float16* vb[4];
#pragma unroll
  for (int dd = 0; dd < 4; ++dd)
    vb[dd] = vtT + ((size_t)b * DD + d0 + dd * 16 + laneM) * SS;

  v8f acc[4] = {};
  for (int k0 = 0; k0 < SS; k0 += 32) {
    f4 f0 = *(const f4*)(arow + k0 + hi * 8);
    f4 f1 = *(const f4*)(arow + k0 + hi * 8 + 4);
    f4 f2 = *(const f4*)(arow + k0 + hi * 8 + 16);
    f4 f3 = *(const f4*)(arow + k0 + hi * 8 + 20);
    v16h A;
#pragma unroll
    for (int j = 0; j < 4; ++j) {
      A[j]      = (_Float16)f0[j];
      A[j + 4]  = (_Float16)f1[j];
      A[j + 8]  = (_Float16)f2[j];
      A[j + 12] = (_Float16)f3[j];
    }
#pragma unroll
    for (int dd = 0; dd < 4; ++dd) {
      h8 b0 = *(const h8*)(vb[dd] + k0 + hi * 8);
      h8 b1 = *(const h8*)(vb[dd] + k0 + hi * 8 + 16);
      const v16h Bm = pack16(b0, b1);
      acc[dd] = __builtin_amdgcn_wmma_f32_16x16x32_f16(
          false, A, false, Bm, (short)0, acc[dd], false, false);
    }
  }
#pragma unroll
  for (int dd = 0; dd < 4; ++dd) {
#pragma unroll
    for (int r = 0; r < 8; ++r) {
      const int M = r + hi * 8;
      out[((size_t)b * SS + srow + M) * DD + d0 + dd * 16 + laneM] = acc[dd][r];
    }
  }
}

// ---------------------------------------------------------------------------
// Kernel 6: exp-map at origin (in place on out): x *= tanh(||x||)/max(||x||,eps)
// ---------------------------------------------------------------------------
__global__ void expmap_kernel(float* __restrict__ out) {
  const int row = blockIdx.x;
  const int tid = threadIdx.x;
  float* p = out + (size_t)row * DD;
  float x0 = p[tid], x1 = p[tid + 256];
  __shared__ float red[256];
  red[tid] = x0 * x0 + x1 * x1;
  __syncthreads();
  for (int s2 = 128; s2 > 0; s2 >>= 1) {
    if (tid < s2) red[tid] += red[tid + s2];
    __syncthreads();
  }
  const float n     = fmaxf(sqrtf(red[0]), EPSF);
  const float scale = tanhf(n) / n;
  p[tid] = x0 * scale;
  p[tid + 256] = x1 * scale;
}

// ---------------------------------------------------------------------------
extern "C" void kernel_launch(void* const* d_in, const int* in_sizes, int n_in,
                              void* d_out, int out_size, void* d_ws, size_t ws_size,
                              hipStream_t stream) {
  const float* q    = (const float*)d_in[0];
  const float* k    = (const float*)d_in[1];
  const float* v    = (const float*)d_in[2];
  const float* beta = (const float*)d_in[3];
  const float* c    = (const float*)d_in[4];

  float* out  = (float*)d_out;                          // [B,S,D] out_hyp
  float* attn = out + (size_t)BB * SS * DD;             // [B,S,S] attn

  // workspace: q_h | k_h | vtT (f16, 8MB each) | qsq | ksq | vscale
  _Float16* qh   = (_Float16*)d_ws;
  _Float16* kh   = qh + (size_t)BB * SS * DD;
  _Float16* vtT  = kh + (size_t)BB * SS * DD;
  float*    qsq  = (float*)(vtT + (size_t)BB * SS * DD);
  float*    ksq  = qsq + (size_t)BB * SS;
  float*    vsc  = ksq + (size_t)BB * SS;

  prep_kernel<<<BB * SS, 256, 0, stream>>>(q, k, v, qh, kh, qsq, ksq, vsc);

  vtan_transpose_kernel<<<BB * (SS / 32) * (DD / 32), 256, 0, stream>>>(v, vsc, vtT);

  // scores: 8 s-tiles per block (one per wave), 4 t-tiles per wave
  const int scoreBlocks = BB * ((SS / 16) / 8) * ((SS / 16) / 4);   // 2048
  scores_kernel<<<scoreBlocks, 256, 0, stream>>>(qh, kh, qsq, ksq, beta, c, attn);

  softmax_kernel<<<BB * SS, 256, 0, stream>>>(attn);

  // av: 8 s-tiles per block, 4 d-tiles per wave
  const int avBlocks = BB * ((SS / 16) / 8) * ((DD / 16) / 4);      // 512
  av_kernel<<<avBlocks, 256, 0, stream>>>(attn, vtT, out);

  expmap_kernel<<<BB * SS, 256, 0, stream>>>(out);
}
